// TernaryLinear_61624190763534
// MI455X (gfx1250) — compile-verified, matrix operand order checked by reading, not run on hardware
//
#include <hip/hip_runtime.h>

typedef __bf16 bf16;
typedef __attribute__((ext_vector_type(16))) __bf16 v16bf;
typedef __attribute__((ext_vector_type(8)))  __bf16 v8bf;
typedef __attribute__((ext_vector_type(8)))  float  v8f;
typedef __attribute__((ext_vector_type(4)))  int    v4i;

#define TILE_M 128
#define TILE_N 256
#define TILE_K 64
#define LDS_PAD 8                       // 16 bytes of row padding
#define LDS_STRIDE (TILE_K + LDS_PAD)   // 72 bf16 = 144 B per row (16B aligned)

// ---- gfx1250 async global->LDS copy (ASYNCcnt path), with safe fallback ----
#if __has_builtin(__builtin_amdgcn_global_load_async_to_lds_b128)
#define ASYNC_CP_B128(gsrc, ldst)                                             \
    __builtin_amdgcn_global_load_async_to_lds_b128((v4i*)(gsrc),              \
                                                   (v4i*)(ldst), 0, 0)
#define HAVE_ASYNC 1
#else
#define ASYNC_CP_B128(gsrc, ldst)                                             \
    do { *(uint4*)(ldst) = *(const uint4*)(gsrc); } while (0)
#define HAVE_ASYNC 0
#endif

#if __has_builtin(__builtin_amdgcn_s_wait_asynccnt)
#define WAIT_ASYNC() __builtin_amdgcn_s_wait_asynccnt(0)
#elif HAVE_ASYNC
#define WAIT_ASYNC() asm volatile("s_wait_asynccnt 0x0" ::: "memory")
#else
#define WAIT_ASYNC() do { } while (0)
#endif

// ---------------------------------------------------------------------------
// Pass 1: ternarize weight  W[4096,4096] f32 -> {-1,0,+1} as bf16 (exact)
// ---------------------------------------------------------------------------
__global__ __launch_bounds__(256) void ternarize_w_kernel(
    const float* __restrict__ W, bf16* __restrict__ Wt)
{
    const int i = (blockIdx.x * 256 + threadIdx.x) * 4;
    const float4 v = *(const float4*)(W + i);

    bf16 o[4];
    float r;
    r = rintf(v.x); o[0] = (bf16)((r > 0.f) ? 1.f : ((r < 0.f) ? -1.f : 0.f));
    r = rintf(v.y); o[1] = (bf16)((r > 0.f) ? 1.f : ((r < 0.f) ? -1.f : 0.f));
    r = rintf(v.z); o[2] = (bf16)((r > 0.f) ? 1.f : ((r < 0.f) ? -1.f : 0.f));
    r = rintf(v.w); o[3] = (bf16)((r > 0.f) ? 1.f : ((r < 0.f) ? -1.f : 0.f));

    *(uint2*)(Wt + i) = *(const uint2*)o;
}

// ---------------------------------------------------------------------------
// Pass 2: Out[M,N] = X[M,K] * Wt[N,K]^T  (bf16 WMMA, f32 accumulate)
//   block: 256 threads = 8 waves (2 M x 4 N); block tile 128x256
//   wave tile 64x64 = 4x4 WMMA accumulators; K staged 64-wide, double buffered
//   B tile: async global->LDS copies (no conversion, no VGPR staging)
//   A tile: global->VGPR f32 prefetch, cvt->bf16, ds_store (pipelined)
// ---------------------------------------------------------------------------
__global__ __launch_bounds__(256) void ternary_gemm_wmma_kernel(
    const float* __restrict__ X, const bf16* __restrict__ Wt,
    float* __restrict__ Out, int M, int N, int K)
{
    __shared__ bf16 Als[2][TILE_M * LDS_STRIDE];   // 2 * 18432 B
    __shared__ bf16 Bls[2][TILE_N * LDS_STRIDE];   // 2 * 36864 B   (~110 KB total)

    const int tid  = threadIdx.x;
    const int lane = tid & 31;
    const int wave = tid >> 5;
    const int wm   = wave & 1;      // 0..1 -> 64 rows
    const int wn   = wave >> 1;     // 0..3 -> 64 cols
    const int half = lane >> 4;     // lane group: 0 = lanes 0-15, 1 = 16-31
    const int l16  = lane & 15;

    const int m_blk = blockIdx.y * TILE_M;
    const int n_blk = blockIdx.x * TILE_N;
    const size_t Ks = (size_t)K;

    v8f acc[4][4] = {};

    // ---------------- prologue: fill stage 0 ----------------
    {
#pragma unroll
        for (int i = 0; i < 8; ++i) {            // B: 256 rows x 8 uint4
            const int j = tid + i * 256;
            const int row = j >> 3, c8 = j & 7;
            ASYNC_CP_B128(Wt + (size_t)(n_blk + row) * Ks + c8 * 8,
                          &Bls[0][row * LDS_STRIDE + c8 * 8]);
        }
#pragma unroll
        for (int i = 0; i < 8; ++i) {            // A: 128 rows x 16 float4
            const int j = tid + i * 256;
            const int row = j >> 4, c4 = j & 15;
            const float4 v = ((const float4*)(X + (size_t)(m_blk + row) * Ks))[c4];
            bf16* d = &Als[0][row * LDS_STRIDE + c4 * 4];
            d[0] = (bf16)v.x; d[1] = (bf16)v.y; d[2] = (bf16)v.z; d[3] = (bf16)v.w;
        }
        WAIT_ASYNC();
        __syncthreads();
    }

    int buf = 0;
    for (int k0 = 0; k0 < K; k0 += TILE_K) {
        const int  nbuf     = buf ^ 1;
        const bool has_next = (k0 + TILE_K) < K;
        const int  kn       = k0 + TILE_K;

        // -------- prefetch next stage (overlaps with WMMA below) --------
        float4 areg[8];
        if (has_next) {
#pragma unroll
            for (int i = 0; i < 8; ++i) {        // B -> LDS, async
                const int j = tid + i * 256;
                const int row = j >> 3, c8 = j & 7;
                ASYNC_CP_B128(Wt + (size_t)(n_blk + row) * Ks + kn + c8 * 8,
                              &Bls[nbuf][row * LDS_STRIDE + c8 * 8]);
            }
#pragma unroll
            for (int i = 0; i < 8; ++i) {        // A f32 -> VGPRs
                const int j = tid + i * 256;
                const int row = j >> 4, c4 = j & 15;
                areg[i] = ((const float4*)(X + (size_t)(m_blk + row) * Ks + kn))[c4];
            }
        }

        // -------- compute current stage: 32 WMMAs per wave --------
#pragma unroll
        for (int kk = 0; kk < TILE_K; kk += 32) {
            // B fragments: lane = col n (l16); halves 0-7: K=kk+half*8+0..7,
            // halves 8-15: K=kk+16+half*8+0..7
            v16bf bfrag[4];
#pragma unroll
            for (int j = 0; j < 4; ++j) {
                const bf16* p =
                    &Bls[buf][(wn * 64 + j * 16 + l16) * LDS_STRIDE + kk + half * 8];
                const v8bf lo = *(const v8bf*)p;
                const v8bf hi = *(const v8bf*)(p + 16);
#pragma unroll
                for (int e = 0; e < 8; ++e) { bfrag[j][e] = lo[e]; bfrag[j][e + 8] = hi[e]; }
            }
#pragma unroll
            for (int i = 0; i < 4; ++i) {
                const bf16* p =
                    &Als[buf][(wm * 64 + i * 16 + l16) * LDS_STRIDE + kk + half * 8];
                const v8bf lo = *(const v8bf*)p;
                const v8bf hi = *(const v8bf*)(p + 16);
                v16bf afrag;
#pragma unroll
                for (int e = 0; e < 8; ++e) { afrag[e] = lo[e]; afrag[e + 8] = hi[e]; }
#pragma unroll
                for (int j = 0; j < 4; ++j)
                    acc[i][j] = __builtin_amdgcn_wmma_f32_16x16x32_bf16(
                        false, afrag, false, bfrag[j], (short)0, acc[i][j],
                        false, false);
            }
        }

        // -------- finish next-stage A fill: cvt + ds_store --------
        if (has_next) {
#pragma unroll
            for (int i = 0; i < 8; ++i) {
                const int j = tid + i * 256;
                const int row = j >> 4, c4 = j & 15;
                bf16* d = &Als[nbuf][row * LDS_STRIDE + c4 * 4];
                d[0] = (bf16)areg[i].x; d[1] = (bf16)areg[i].y;
                d[2] = (bf16)areg[i].z; d[3] = (bf16)areg[i].w;
            }
        }
        WAIT_ASYNC();
        __syncthreads();
        buf = nbuf;
    }

    // -------- epilogue: D layout -> VGPR r holds M=r+8*half, N=l16 --------
#pragma unroll
    for (int i = 0; i < 4; ++i) {
        const int mrow = m_blk + wm * 64 + i * 16 + half * 8;
#pragma unroll
        for (int j = 0; j < 4; ++j) {
            const int ncol = n_blk + wn * 64 + j * 16 + l16;
            float* op = Out + (size_t)mrow * N + ncol;   // hoisted base
#pragma unroll
            for (int r = 0; r < 8; ++r)
                op[(size_t)r * N] = acc[i][j][r];        // immediate offsets
        }
    }
}

// ---------------------------------------------------------------------------
extern "C" void kernel_launch(void* const* d_in, const int* in_sizes, int n_in,
                              void* d_out, int out_size, void* d_ws, size_t ws_size,
                              hipStream_t stream)
{
    const float* x = (const float*)d_in[0];   // [8192, 4096] f32
    const float* w = (const float*)d_in[1];   // [4096, 4096] f32
    float* out     = (float*)d_out;           // [8192, 4096] f32
    bf16*  wt      = (bf16*)d_ws;             // 32 MB ternary-bf16 weight

    const int M = 8192, N = 4096, K = 4096;

    ternarize_w_kernel<<<(N * K) / 1024, 256, 0, stream>>>(w, wt);

    dim3 grid(N / TILE_N, M / TILE_M);
    ternary_gemm_wmma_kernel<<<grid, 256, 0, stream>>>(x, wt, out, M, N, K);
}
